// SpeTB_30571577213037
// MI455X (gfx1250) — matrix operand order repaired and checked by reading, not compiled, not run on hardware
//
#include <hip/hip_runtime.h>
#include <math.h>

// ---------------- problem constants ----------------
#define IMG   128
#define NB    4
#define DIM   180
#define KP    192                 // DIM padded to WMMA tile multiple
#define NTOK  (NB*IMG*IMG)        // 65536
#define HEADS 6
#define HD    30                  // head dim
#define WSZ   16
#define OWW   24                  // overlapped window
#define PADW  4
#define SCALEQ 0.18257418583505536f   // 30^-0.5

typedef __attribute__((ext_vector_type(16))) __bf16 v16bf;
typedef __attribute__((ext_vector_type(8)))  __bf16 v8bf;
typedef __attribute__((ext_vector_type(2)))  __bf16 v2bf;
typedef __attribute__((ext_vector_type(8)))  float  v8f;

static __device__ __forceinline__ unsigned bf16bits(float f) {
  unsigned u = __builtin_bit_cast(unsigned, f);
  return (u + 0x7FFFu + ((u >> 16) & 1u)) >> 16;   // RNE
}
static __device__ __forceinline__ __bf16 f2bf(float f) {
  unsigned short s = (unsigned short)bf16bits(f);
  return __builtin_bit_cast(__bf16, s);
}

// 16-bit A-matrix VGPR layout (ISA 7.12.2): vgpr v holds K pair starting at:
static __device__ __forceinline__ int akidx(int v, int h) {
  return (v < 4) ? (2 * v + 8 * h) : (16 + 2 * (v - 4) + 8 * h);
}

static __device__ __forceinline__ v8f wmma_bf16(v16bf a, v16bf b, v8f c) {
  return __builtin_amdgcn_wmma_f32_16x16x32_bf16(false, a, false, b, (short)0, c,
                                                 false, false);
}

// assemble a 16-element fragment from two 16B runs (ISA layout: elements 0..7
// are K k0+8h..+7, elements 8..15 are K k0+16+8h..+7 for A; analogous for B)
static __device__ __forceinline__ v16bf frag_from2(uint4 a, uint4 b) {
  v8bf lo = __builtin_bit_cast(v8bf, a), hi = __builtin_bit_cast(v8bf, b);
  v16bf r;
#pragma unroll
  for (int i = 0; i < 8; ++i) { r[i] = lo[i]; r[8 + i] = hi[i]; }
  return r;
}
static __device__ __forceinline__ v16bf frag_zero() {
  v16bf z;
#pragma unroll
  for (int i = 0; i < 16; ++i) z[i] = __builtin_bit_cast(__bf16, (unsigned short)0);
  return z;
}

// A tile (16x32) fragment from row-major bf16 buffer: two b128 loads per lane.
static __device__ __forceinline__ v16bf load_a(const __bf16* base, int m0, int ld, int k0) {
  int lane = threadIdx.x & 31;
  int m = m0 + (lane & 15), h = lane >> 4;
  const __bf16* p = base + (size_t)m * ld + k0 + 8 * h;
  return frag_from2(*(const uint4*)p, *(const uint4*)(p + 16));
}

// A tile with horizontally shifted pixel (conv), zero outside [0,IMG).
static __device__ __forceinline__ v16bf load_a_shift(const __bf16* sr, int ld, int m0,
                                                     int dx, int k0) {
  int lane = threadIdx.x & 31;
  int pc = m0 + (lane & 15) + dx;
  bool ok = (pc >= 0 && pc < IMG);
  int pcs = ok ? pc : 0;
  int h = lane >> 4;
  const __bf16* p = sr + (size_t)pcs * ld + k0 + 8 * h;
  v16bf a = frag_from2(*(const uint4*)p, *(const uint4*)(p + 16));
  return ok ? a : frag_zero();
}

// B tile (32x16) fragment from packed pairs P[n*ldk2 + kp] (k-major per out-ch):
// two contiguous uint4 loads per lane.
static __device__ __forceinline__ v16bf load_bp(const unsigned* __restrict__ P,
                                                int n0, int k0, int ldk2) {
  int lane = threadIdx.x & 31;
  int n = n0 + (lane & 15), h = lane >> 4;
  const uint4* p = (const uint4*)(P + (size_t)n * ldk2 + (k0 >> 1) + 4 * h);
  return frag_from2(p[0], p[2]);
}

// ------------- async global->LDS staging (CDNA5, ASYNCcnt) -------------
static __device__ __forceinline__ void async_copy_b128(unsigned lds_byte_off,
                                                       const void* gsrc, int n128,
                                                       int tid, int nthr) {
  unsigned long long g0 = (unsigned long long)(size_t)gsrc;
  for (int i = tid; i < n128; i += nthr) {
    unsigned lo = lds_byte_off + (unsigned)i * 16u;
    unsigned long long ga = g0 + (unsigned long long)i * 16u;
    asm volatile("global_load_async_to_lds_b128 %0, %1, off"
                 :: "v"(lo), "v"(ga) : "memory");
  }
}
static __device__ __forceinline__ void async_wait_all() {
  asm volatile("s_wait_asynccnt 0x0" ::: "memory");
}

// ---------------- kernel 0: init (workspace is poisoned) ----------------
__global__ void k_init(unsigned* __restrict__ ao32, float* __restrict__ pool) {
  size_t n = (size_t)NTOK * KP / 2;
  for (size_t i = (size_t)blockIdx.x * blockDim.x + threadIdx.x; i < n;
       i += (size_t)gridDim.x * blockDim.x) ao32[i] = 0u;
  size_t t = (size_t)blockIdx.x * blockDim.x + threadIdx.x;
  if (t < NB * DIM) pool[t] = 0.f;
}

// ------------- weight packers: fp32 -> padded bf16 pairs, k-major per row -------------
// square DIMxDIM weight -> P[n*96 + kp], n<192, kp<96
__global__ void k_packw(const float* __restrict__ W, unsigned* __restrict__ P) {
  int idx = blockIdx.x * 256 + threadIdx.x;
  if (idx >= KP * (KP / 2)) return;
  int n = idx / (KP / 2), kp = idx - n * (KP / 2);
  int k = kp * 2;
  float x0 = (n < DIM && k < DIM)     ? W[(size_t)n * DIM + k]     : 0.f;
  float x1 = (n < DIM && k + 1 < DIM) ? W[(size_t)n * DIM + k + 1] : 0.f;
  P[idx] = (bf16bits(x1) << 16) | (bf16bits(x0) & 0xFFFFu);
}
// conv1 weight (60,180,3,3) -> per-tap [64][96] (out-ch x ci pairs)
__global__ void k_packc1(const float* __restrict__ w1, unsigned* __restrict__ P) {
  int idx = blockIdx.x * 256 + threadIdx.x;
  if (idx >= 9 * 64 * 96) return;
  int tap = idx / (64 * 96); int r = idx - tap * (64 * 96);
  int o = r / 96, kp = r - o * 96;
  int ky = tap / 3, kx = tap - ky * 3;
  int ci = kp * 2;
  float x0 = (o < 60 && ci < DIM)     ? w1[(((size_t)o * DIM + ci) * 3 + ky) * 3 + kx]     : 0.f;
  float x1 = (o < 60 && ci + 1 < DIM) ? w1[(((size_t)o * DIM + ci + 1) * 3 + ky) * 3 + kx] : 0.f;
  P[idx] = (bf16bits(x1) << 16) | (bf16bits(x0) & 0xFFFFu);
}
// conv2 weight (180,60,3,3) -> per-tap [192][32] (out-ch x ci pairs)
__global__ void k_packc2(const float* __restrict__ w2, unsigned* __restrict__ P) {
  int idx = blockIdx.x * 256 + threadIdx.x;
  if (idx >= 9 * 192 * 32) return;
  int tap = idx / (192 * 32); int r = idx - tap * (192 * 32);
  int o = r / 32, kp = r - o * 32;
  int ky = tap / 3, kx = tap - ky * 3;
  int ci = kp * 2;
  float x0 = (o < DIM && ci < 60)     ? w2[(((size_t)o * 60 + ci) * 3 + ky) * 3 + kx]     : 0.f;
  float x1 = (o < DIM && ci + 1 < 60) ? w2[(((size_t)o * 60 + ci + 1) * 3 + ky) * 3 + kx] : 0.f;
  P[idx] = (bf16bits(x1) << 16) | (bf16bits(x0) & 0xFFFFu);
}

// ---------------- kernel 1: LN1 + fused Q/K/V GEMM ----------------
__global__ __launch_bounds__(256) void k_ln_qkv(
    const float* __restrict__ x, const float* __restrict__ y,
    const float* __restrict__ g1, const float* __restrict__ bl1,
    const unsigned* __restrict__ pkq, const unsigned* __restrict__ pkk,
    const unsigned* __restrict__ pkv,
    const float* __restrict__ bq, const float* __restrict__ bk,
    const float* __restrict__ bv,
    __bf16* __restrict__ qb, __bf16* __restrict__ kb, __bf16* __restrict__ vb) {
  extern __shared__ char smem[];
  __bf16* sx = (__bf16*)smem;            // [128][KP]
  __bf16* sy = sx + 128 * KP;
  const int tok0 = blockIdx.x * 128;
  const int t = threadIdx.x;
  {
    int r = t & 127;
    const float* src = (t < 128) ? x + (size_t)(tok0 + r) * DIM
                                 : y + (size_t)(tok0 + r) * DIM;
    __bf16* dst = (t < 128) ? sx + r * KP : sy + r * KP;
    float m = 0.f;
    for (int i = 0; i < DIM; ++i) m += src[i];
    m *= (1.f / DIM);
    float var = 0.f;
    for (int i = 0; i < DIM; ++i) { float d = src[i] - m; var += d * d; }
    float inv = rsqrtf(var * (1.f / DIM) + 1e-5f);
    for (int i = 0; i < DIM; ++i) dst[i] = f2bf((src[i] - m) * inv * g1[i] + bl1[i]);
    for (int i = DIM; i < KP; ++i) dst[i] = f2bf(0.f);
  }
  __syncthreads();
  const int wave = t >> 5, lane = t & 31;
  const int nloc = lane & 15, h = lane >> 4;
#pragma unroll 1
  for (int o = 0; o < 3; ++o) {
    const __bf16* A    = (o == 0) ? sx : sy;          // Q from xn; K,V from yn
    const unsigned* P  = (o == 0) ? pkq : (o == 1) ? pkk : pkv;
    const float* bias  = (o == 0) ? bq : (o == 1) ? bk : bv;
    __bf16* out        = (o == 0) ? qb : (o == 1) ? kb : vb;
    const float sc = (o == 0) ? SCALEQ : 1.f;         // fold attention scale into Q
#pragma unroll 1
    for (int j = 0; j < 12; ++j) {
      v8f acc = {};
#pragma unroll
      for (int kk = 0; kk < 6; ++kk) {
        if (kk < 5)
          __builtin_prefetch(P + (size_t)(j * 16) * (KP / 2) + ((kk * 32 + 32) >> 1), 0, 1);
        v16bf af = load_a(A, wave * 16, KP, kk * 32);
        v16bf bf = load_bp(P, j * 16, kk * 32, KP / 2);
        acc = wmma_bf16(af, bf, acc);
      }
      int n = j * 16 + nloc;
      float bb = bias[n < DIM ? n : 0];
      bb = (n < DIM) ? bb : 0.f;
#pragma unroll
      for (int r = 0; r < 8; ++r) {
        int tok = tok0 + wave * 16 + r + 8 * h;       // C/D: m = r + 8*h
        float val = (acc[r] + bb) * sc;
        out[(size_t)tok * KP + n] = f2bf(n < DIM ? val : 0.f);
      }
    }
  }
}

// ---------------- kernel 2: windowed channel attention ----------------
__global__ __launch_bounds__(32) void k_attn(
    const __bf16* __restrict__ qb, const __bf16* __restrict__ kb,
    const __bf16* __restrict__ vb, __bf16* __restrict__ aout) {
  __shared__ float  sattn[32 * 32];
  __shared__ __bf16 sattnb[32 * 32];
  const int head = blockIdx.x >> 8;
  const int w = blockIdx.x & 255;
  const int b = w >> 6, win = w & 63, wr = win >> 3, wc = win & 7;
  const int lane = threadIdx.x, col = lane & 15, h = lane >> 4;
  const size_t base = (size_t)b * (IMG * IMG);
  const __bf16 BZ = f2bf(0.f);

  v8f acc[2][2];
#pragma unroll
  for (int mi = 0; mi < 2; ++mi)
#pragma unroll
    for (int ej = 0; ej < 2; ++ej) acc[mi][ej] = {};

#pragma unroll 1
  for (int nn = 0; nn < OWW * OWW; nn += 32) {
    v16bf af[2], bfr[2];
#pragma unroll
    for (int mi = 0; mi < 2; ++mi) {
      int d = mi * 16 + col;                           // A[m=d][k=n] = q[n,d]
#pragma unroll
      for (int v = 0; v < 8; ++v) {
        int n0 = nn + akidx(v, h);
#pragma unroll
        for (int u = 0; u < 2; ++u) {
          int n = n0 + u;
          int rr = n / OWW, cc = n - rr * OWW;
          int gr = wr * WSZ + rr - PADW, gc = wc * WSZ + cc - PADW;
          bool ok = (gr >= 0) && (gr < IMG) && (gc >= 0) && (gc < IMG);
          size_t tok = base + (ok ? (gr * IMG + gc) : 0);
          __bf16 val = qb[tok * KP + head * HD + d];
          af[mi][2 * v + u] = ok ? val : BZ;
        }
      }
    }
#pragma unroll
    for (int ej = 0; ej < 2; ++ej) {
      int e = ej * 16 + col;                           // B[k=n][n=e] = k[n,e]
#pragma unroll
      for (int v = 0; v < 8; ++v) {
        int k0 = nn + akidx(v, h);
#pragma unroll
        for (int u = 0; u < 2; ++u) {
          int n = k0 + u;
          int rr = n / OWW, cc = n - rr * OWW;
          int gr = wr * WSZ + rr - PADW, gc = wc * WSZ + cc - PADW;
          bool ok = (gr >= 0) && (gr < IMG) && (gc >= 0) && (gc < IMG);
          size_t tok = base + (ok ? (gr * IMG + gc) : 0);
          __bf16 val = kb[tok * KP + head * HD + e];
          bfr[ej][2 * v + u] = ok ? val : BZ;
        }
      }
    }
#pragma unroll
    for (int mi = 0; mi < 2; ++mi)
#pragma unroll
      for (int ej = 0; ej < 2; ++ej)
        acc[mi][ej] = wmma_bf16(af[mi], bfr[ej], acc[mi][ej]);
  }
  // spill attn -> LDS (C/D layout: m = r + 8*h, n = col)
#pragma unroll
  for (int mi = 0; mi < 2; ++mi)
#pragma unroll
    for (int ej = 0; ej < 2; ++ej)
#pragma unroll
      for (int r = 0; r < 8; ++r)
        sattn[(mi * 16 + r + 8 * h) * 32 + ej * 16 + col] = acc[mi][ej][r];
  __syncthreads();
  // softmax over e (only e<30 valid; padded rows/cols zeroed for stage 2)
  {
    int d = lane;
    if (d < HD) {
      float mx = -1e30f;
      for (int e = 0; e < HD; ++e) mx = fmaxf(mx, sattn[d * 32 + e]);
      float s = 0.f;
      for (int e = 0; e < HD; ++e) s += expf(sattn[d * 32 + e] - mx);
      float inv = 1.f / s;
      for (int e = 0; e < HD; ++e) sattnb[d * 32 + e] = f2bf(expf(sattn[d * 32 + e] - mx) * inv);
      for (int e = HD; e < 32; ++e) sattnb[d * 32 + e] = BZ;
    } else {
      for (int e = 0; e < 32; ++e) sattnb[d * 32 + e] = BZ;
    }
  }
  __syncthreads();
  // stage 2: out[q,e] = sum_d v[q,d] * attn[d,e]  (M=256, N=32, K=32)
  v16bf bf2[2];
#pragma unroll
  for (int ej = 0; ej < 2; ++ej) {
    int e = ej * 16 + col;
#pragma unroll
    for (int v = 0; v < 8; ++v) {
      int k0 = akidx(v, h);
      bf2[ej][2 * v]     = sattnb[k0 * 32 + e];
      bf2[ej][2 * v + 1] = sattnb[(k0 + 1) * 32 + e];
    }
  }
  v8f vz = {};
#pragma unroll 1
  for (int qi = 0; qi < 16; ++qi) {
    int qpos = qi * 16 + col;
    size_t tokA = base + (size_t)(wr * WSZ + (qpos >> 4)) * IMG + wc * WSZ + (qpos & 15);
    v16bf af2;
#pragma unroll
    for (int v = 0; v < 8; ++v) {
      int k0 = akidx(v, h);
      v2bf p = __builtin_bit_cast(v2bf, *(const unsigned*)(vb + tokA * KP + head * HD + k0));
      af2[2 * v] = p[0]; af2[2 * v + 1] = p[1];
    }
#pragma unroll
    for (int ej = 0; ej < 2; ++ej) {
      v8f o = wmma_bf16(af2, bf2[ej], vz);
      int e = ej * 16 + col;
      if (e < HD) {
#pragma unroll
        for (int r = 0; r < 8; ++r) {
          int qp = qi * 16 + r + 8 * h;
          size_t tok = base + (size_t)(wr * WSZ + (qp >> 4)) * IMG + wc * WSZ + (qp & 15);
          aout[tok * KP + head * HD + e] = f2bf(o[r]);
        }
      }
    }
  }
}

// ---------------- kernel 3: output projection + residual ----------------
__global__ __launch_bounds__(256) void k_proj(
    const __bf16* __restrict__ aout, const unsigned* __restrict__ pkp,
    const float* __restrict__ bp, const float* __restrict__ yres,
    float* __restrict__ x2) {
  extern __shared__ char smem[];
  __bf16* sA = (__bf16*)smem;                // [128][KP]
  const int tok0 = blockIdx.x * 128;
  {
    unsigned lbase = __builtin_amdgcn_groupstaticsize();
    async_copy_b128(lbase, aout + (size_t)tok0 * KP, 128 * KP * 2 / 16,
                    threadIdx.x, 256);
    async_wait_all();
  }
  __syncthreads();
  const int wave = threadIdx.x >> 5, lane = threadIdx.x & 31;
  const int nloc = lane & 15, h = lane >> 4;
#pragma unroll 1
  for (int j = 0; j < 12; ++j) {
    v8f acc = {};
#pragma unroll
    for (int kk = 0; kk < 6; ++kk) {
      v16bf af = load_a(sA, wave * 16, KP, kk * 32);
      v16bf bf = load_bp(pkp, j * 16, kk * 32, KP / 2);
      acc = wmma_bf16(af, bf, acc);
    }
    int n = j * 16 + nloc;
    if (n < DIM) {
      float bb = bp[n];
#pragma unroll
      for (int r = 0; r < 8; ++r) {
        size_t tok = (size_t)tok0 + wave * 16 + r + 8 * h;
        x2[tok * DIM + n] = acc[r] + bb + yres[tok * DIM + n];
      }
    }
  }
}

// ---------------- kernel 4: LN2 -> bf16 NHWC (conv input) ----------------
__global__ __launch_bounds__(256) void k_ln2(
    const float* __restrict__ x2, const float* __restrict__ g,
    const float* __restrict__ bb, __bf16* __restrict__ xn2) {
  size_t tok = (size_t)blockIdx.x * 256 + threadIdx.x;
  if (tok >= NTOK) return;
  const float* src = x2 + tok * DIM;
  float m = 0.f;
  for (int i = 0; i < DIM; ++i) m += src[i];
  m *= (1.f / DIM);
  float var = 0.f;
  for (int i = 0; i < DIM; ++i) { float d = src[i] - m; var += d * d; }
  float inv = rsqrtf(var * (1.f / DIM) + 1e-5f);
  __bf16* dst = xn2 + tok * KP;
  for (int i = 0; i < DIM; ++i) dst[i] = f2bf((src[i] - m) * inv * g[i] + bb[i]);
  for (int i = DIM; i < KP; ++i) dst[i] = f2bf(0.f);
}

// ---------------- kernel 5: conv3x3 180->60 + GELU (implicit GEMM) ----------------
__global__ __launch_bounds__(256) void k_conv1(
    const __bf16* __restrict__ xn2, const unsigned* __restrict__ pc1,
    const float* __restrict__ b1, __bf16* __restrict__ t1) {
  extern __shared__ char smem[];
  __bf16* srow = (__bf16*)smem;              // [3][128][KP]
  const int b = blockIdx.x >> 7, gr = blockIdx.x & 127;
  {
    unsigned lbase = __builtin_amdgcn_groupstaticsize();
#pragma unroll 1
    for (int rr = 0; rr < 3; ++rr) {
      int g = gr + rr - 1;
      if (g >= 0 && g < IMG) {
        async_copy_b128(lbase + rr * 128 * KP * 2,
                        xn2 + ((size_t)b * IMG * IMG + (size_t)g * IMG) * KP,
                        128 * KP * 2 / 16, threadIdx.x, 256);
      } else {
        uint4 z = {0u, 0u, 0u, 0u};
        uint4* d = (uint4*)(srow + rr * 128 * KP);
        for (int i = threadIdx.x; i < 128 * KP / 8; i += 256) d[i] = z;
      }
    }
    async_wait_all();
  }
  __syncthreads();
  const int wave = threadIdx.x >> 5, lane = threadIdx.x & 31;
  const int oloc = lane & 15;
  const int h = lane >> 4;
#pragma unroll 1
  for (int j = 0; j < 4; ++j) {
    v8f acc = {};
#pragma unroll 1
    for (int ky = 0; ky < 3; ++ky) {
      const __bf16* sr = srow + ky * 128 * KP;
#pragma unroll 1
      for (int kx = 0; kx < 3; ++kx) {
        const unsigned* Pt = pc1 + (ky * 3 + kx) * (64 * 96);
#pragma unroll
        for (int kk = 0; kk < 6; ++kk) {
          v16bf af = load_a_shift(sr, KP, wave * 16, kx - 1, kk * 32);
          v16bf bf = load_bp(Pt, j * 16, kk * 32, 96);
          acc = wmma_bf16(af, bf, acc);
        }
      }
    }
    int o = j * 16 + oloc;
    float bb = b1[o < 60 ? o : 0];
    bb = (o < 60) ? bb : 0.f;
#pragma unroll
    for (int r = 0; r < 8; ++r) {
      int pc = wave * 16 + r + 8 * h;
      size_t tok = (size_t)b * IMG * IMG + (size_t)gr * IMG + pc;
      float vv = acc[r] + bb;
      float gl = 0.5f * vv * (1.f + erff(vv * 0.70710678118f));  // exact GELU
      t1[tok * 64 + o] = f2bf(o < 60 ? gl : 0.f);                // pad 60..63 = 0
    }
  }
}

// ---------------- kernel 6: conv3x3 60->180 + channel pool ----------------
__global__ __launch_bounds__(256) void k_conv2(
    const __bf16* __restrict__ t1, const unsigned* __restrict__ pc2,
    const float* __restrict__ b2, float* __restrict__ t2,
    float* __restrict__ pool) {
  extern __shared__ char smem[];
  __bf16* srow = (__bf16*)smem;              // [3][128][64]
  __shared__ float spool[KP];
  const int b = blockIdx.x >> 7, gr = blockIdx.x & 127;
  for (int i = threadIdx.x; i < KP; i += 256) spool[i] = 0.f;
  {
    unsigned lbase = __builtin_amdgcn_groupstaticsize();
#pragma unroll 1
    for (int rr = 0; rr < 3; ++rr) {
      int g = gr + rr - 1;
      if (g >= 0 && g < IMG) {
        async_copy_b128(lbase + rr * 128 * 64 * 2,
                        t1 + ((size_t)b * IMG * IMG + (size_t)g * IMG) * 64,
                        128 * 64 * 2 / 16, threadIdx.x, 256);
      } else {
        uint4 z = {0u, 0u, 0u, 0u};
        uint4* d = (uint4*)(srow + rr * 128 * 64);
        for (int i = threadIdx.x; i < 128 * 64 / 8; i += 256) d[i] = z;
      }
    }
    async_wait_all();
  }
  __syncthreads();
  const int wave = threadIdx.x >> 5, lane = threadIdx.x & 31;
  const int oloc = lane & 15;
  const int h = lane >> 4;
#pragma unroll 1
  for (int j = 0; j < 12; ++j) {
    v8f acc = {};
#pragma unroll 1
    for (int ky = 0; ky < 3; ++ky) {
      const __bf16* sr = srow + ky * 128 * 64;
#pragma unroll 1
      for (int kx = 0; kx < 3; ++kx) {
        const unsigned* Pt = pc2 + (ky * 3 + kx) * (192 * 32);
#pragma unroll
        for (int kk = 0; kk < 2; ++kk) {
          v16bf af = load_a_shift(sr, 64, wave * 16, kx - 1, kk * 32);
          v16bf bf = load_bp(Pt, j * 16, kk * 32, 32);
          acc = wmma_bf16(af, bf, acc);
        }
      }
    }
    int o = j * 16 + oloc;
    if (o < DIM) {
      float bb = b2[o];
#pragma unroll
      for (int r = 0; r < 8; ++r) {
        int pc = wave * 16 + r + 8 * h;
        size_t tok = (size_t)b * IMG * IMG + (size_t)gr * IMG + pc;
        float vv = acc[r] + bb;
        t2[tok * DIM + o] = vv;
        atomicAdd(&spool[o], vv);        // ds_add_f32 block-level pool
      }
    }
  }
  __syncthreads();
  for (int i = threadIdx.x; i < DIM; i += 256)
    atomicAdd(&pool[b * DIM + i], spool[i]);   // global_atomic_add_f32
}

// ---------------- kernel 7: squeeze-excite gate ----------------
__global__ void k_se(const float* __restrict__ pool,
                     const float* __restrict__ w1, const float* __restrict__ b1,
                     const float* __restrict__ w2, const float* __restrict__ b2,
                     float* __restrict__ amul) {
  int idx = blockIdx.x * blockDim.x + threadIdx.x;
  if (idx >= NB * DIM) return;
  int b = idx / DIM, c = idx - b * DIM;
  const float invHW = 1.f / (IMG * IMG);
  float hid[6];
#pragma unroll
  for (int j = 0; j < 6; ++j) {
    float s = b1[j];
    for (int c2 = 0; c2 < DIM; ++c2) s += pool[b * DIM + c2] * invHW * w1[j * DIM + c2];
    hid[j] = fmaxf(s, 0.f);
  }
  float s2 = b2[c];
#pragma unroll
  for (int j = 0; j < 6; ++j) s2 += hid[j] * w2[c * 6 + j];
  amul[idx] = 1.f / (1.f + expf(-s2));
}

// ---------------- kernel 8: t2 * gate + x2 -> out ----------------
__global__ void k_final(const float* __restrict__ t2, const float* __restrict__ amul,
                        const float* __restrict__ x2, float* __restrict__ out) {
  size_t i = (size_t)blockIdx.x * blockDim.x + threadIdx.x;
  if (i >= (size_t)NTOK * DIM) return;
  size_t tok = i / DIM; int c = (int)(i - tok * DIM);
  int b = (int)(tok >> 14);
  out[i] = t2[i] * amul[b * DIM + c] + x2[i];
}

// ---------------- host-side launch ----------------
extern "C" void kernel_launch(void* const* d_in, const int* in_sizes, int n_in,
                              void* d_out, int out_size, void* d_ws, size_t ws_size,
                              hipStream_t stream) {
  (void)in_sizes; (void)n_in; (void)out_size; (void)ws_size;
  const float* x    = (const float*)d_in[0];
  const float* y    = (const float*)d_in[1];
  const float* n1g  = (const float*)d_in[2];
  const float* n1b  = (const float*)d_in[3];
  const float* wq   = (const float*)d_in[4];
  const float* bq   = (const float*)d_in[5];
  const float* wk   = (const float*)d_in[6];
  const float* bk   = (const float*)d_in[7];
  const float* wv   = (const float*)d_in[8];
  const float* bv   = (const float*)d_in[9];
  const float* wp   = (const float*)d_in[10];
  const float* bp   = (const float*)d_in[11];
  const float* n2g  = (const float*)d_in[12];
  const float* n2b  = (const float*)d_in[13];
  const float* cw1  = (const float*)d_in[14];
  const float* cb1  = (const float*)d_in[15];
  const float* cw2  = (const float*)d_in[16];
  const float* cb2  = (const float*)d_in[17];
  const float* caw1 = (const float*)d_in[18];
  const float* cab1 = (const float*)d_in[19];
  const float* caw2 = (const float*)d_in[20];
  const float* cab2 = (const float*)d_in[21];
  float* outp = (float*)d_out;

  char* wsb = (char*)d_ws;
  size_t off = 0;
  auto take = [&](size_t bytes) {
    char* r = wsb + off;
    off = (off + bytes + 255) & ~(size_t)255;
    return r;
  };
  const size_t SZB = (size_t)NTOK * KP * 2;    // bf16 [NTOK][192]
  const size_t SZF = (size_t)NTOK * DIM * 4;   // f32  [NTOK][180]
  __bf16* qb    = (__bf16*)take(SZB);
  __bf16* kbuf  = (__bf16*)take(SZB);
  __bf16* vbuf  = (__bf16*)take(SZB);
  __bf16* ao    = (__bf16*)take(SZB);
  float*  x2    = (float*)take(SZF);
  __bf16* xn2   = (__bf16*)take(SZB);
  __bf16* t1    = (__bf16*)take((size_t)NTOK * 64 * 2);
  float*  t2    = (float*)take(SZF);
  float*  pool  = (float*)take((size_t)NB * DIM * 4);
  float*  amul  = (float*)take((size_t)NB * DIM * 4);
  unsigned* pkq = (unsigned*)take((size_t)KP * (KP / 2) * 4);
  unsigned* pkk = (unsigned*)take((size_t)KP * (KP / 2) * 4);
  unsigned* pkv = (unsigned*)take((size_t)KP * (KP / 2) * 4);
  unsigned* pkp = (unsigned*)take((size_t)KP * (KP / 2) * 4);
  unsigned* pc1 = (unsigned*)take((size_t)9 * 64 * 96 * 4);
  unsigned* pc2 = (unsigned*)take((size_t)9 * 192 * 32 * 4);

  k_init<<<4096, 256, 0, stream>>>((unsigned*)ao, pool);
  k_packw<<<KP * (KP / 2) / 256, 256, 0, stream>>>(wq, pkq);
  k_packw<<<KP * (KP / 2) / 256, 256, 0, stream>>>(wk, pkk);
  k_packw<<<KP * (KP / 2) / 256, 256, 0, stream>>>(wv, pkv);
  k_packw<<<KP * (KP / 2) / 256, 256, 0, stream>>>(wp, pkp);
  k_packc1<<<9 * 64 * 96 / 256, 256, 0, stream>>>(cw1, pc1);
  k_packc2<<<9 * 192 * 32 / 256, 256, 0, stream>>>(cw2, pc2);
  k_ln_qkv<<<NTOK / 128, 256, 2 * 128 * KP * 2, stream>>>(
      x, y, n1g, n1b, pkq, pkk, pkv, bq, bk, bv, qb, kbuf, vbuf);
  k_attn<<<HEADS * NB * 64, 32, 0, stream>>>(qb, kbuf, vbuf, ao);
  k_proj<<<NTOK / 128, 256, 128 * KP * 2, stream>>>(ao, pkp, bp, y, x2);
  k_ln2<<<NTOK / 256, 256, 0, stream>>>(x2, n2g, n2b, xn2);
  k_conv1<<<NB * IMG, 256, 3 * 128 * KP * 2, stream>>>(xn2, pc1, cb1, t1);
  k_conv2<<<NB * IMG, 256, 3 * 128 * 64 * 2, stream>>>(t1, pc2, cb2, t2, pool);
  k_se<<<(NB * DIM + 255) / 256, 256, 0, stream>>>(pool, caw1, cab1, caw2, cab2, amul);
  k_final<<<(int)(((size_t)NTOK * DIM + 255) / 256), 256, 0, stream>>>(t2, amul, x2, outp);
}
